// Linker_34857954574673
// MI455X (gfx1250) — matrix operand order repaired and checked by reading, not compiled, not run on hardware
//
#include <hip/hip_runtime.h>
#include <hip/hip_bf16.h>
#include <stdint.h>

// ---------------- problem constants (from setup_inputs) ----------------
constexpr int H = 384, W = 512, HW = H * W;
constexpr int MARGIN = 10, STRIDEP = 4;
constexpr int NGX = 123;                // len(range(10, 502, 4))
constexpr int NGY = 91;                 // len(range(10, 374, 4))
constexpr int NGRID = NGX * NGY;        // 11193
constexpr int NSLOTS = 2 * HW / (STRIDEP * STRIDEP); // 24576
constexpr int PR_ROWS = 4;              // rows per prune block (TDM tile height)

typedef __attribute__((ext_vector_type(16))) _Float16 v16h;
typedef __attribute__((ext_vector_type(8)))  float    v8f;
typedef __attribute__((ext_vector_type(4)))  unsigned u32x4;
typedef __attribute__((ext_vector_type(8)))  int      i32x8;
typedef __attribute__((ext_vector_type(4)))  int      i32x4;

// ---------------- bilinear helper (matches pips bilinear_sample2d) ------
struct Bil { int i00, i01, i10, i11; float w00, w01, w10, w11; };
__device__ inline Bil mkbil(float x, float y) {
  Bil b;
  float x0f = floorf(x), y0f = floorf(y);
  float lx = x - x0f, ly = y - y0f;
  int x0 = (int)fminf(fmaxf(x0f,        0.f), (float)(W - 1));
  int x1 = (int)fminf(fmaxf(x0f + 1.f,  0.f), (float)(W - 1));
  int y0 = (int)fminf(fmaxf(y0f,        0.f), (float)(H - 1));
  int y1 = (int)fminf(fmaxf(y0f + 1.f,  0.f), (float)(H - 1));
  b.w00 = (1.f - lx) * (1.f - ly);  b.w01 = lx * (1.f - ly);
  b.w10 = (1.f - lx) * ly;          b.w11 = lx * ly;
  b.i00 = y0 * W + x0;  b.i01 = y0 * W + x1;
  b.i10 = y1 * W + x0;  b.i11 = y1 * W + x1;
  return b;
}

// ---------------- TDM: load a PR_ROWS x W f32 tile into LDS ------------
__device__ inline void tdm_load_rows(const void* gptr, unsigned lds_off) {
  unsigned long long ga = (unsigned long long)(uintptr_t)gptr;
  u32x4 g0;
  g0.x = 1u;                                            // count=1, no gather
  g0.y = lds_off;                                       // lds_addr (bytes)
  g0.z = (unsigned)(ga & 0xFFFFFFFFu);                  // global_addr[31:0]
  g0.w = (unsigned)((ga >> 32) & 0x01FFFFFFu) | (2u << 30); // addr[56:32], type=2
  i32x8 g1;
  g1[0] = (2 << 16);                                    // data_size = 4B
  g1[1] = (int)((W & 0xFFFF) << 16);                    // tensor_dim0 lo16
  g1[2] = (int)(((unsigned)H & 0xFFFFu) << 16);         // dim0 hi16=0 | tensor_dim1 lo16
  g1[3] = (int)(((unsigned)W & 0xFFFFu) << 16);         // dim1 hi16=0 | tile_dim0 = W
  g1[4] = PR_ROWS;                                      // tile_dim1 | tile_dim2=0
  g1[5] = W;                                            // tensor_dim0_stride lo32
  g1[6] = 0;                                            // stride hi | dim1_stride lo
  g1[7] = 0;
  i32x4 z4 = {0, 0, 0, 0};
#if defined(__clang_major__) && __clang_major__ >= 23
  i32x8 z8 = {0, 0, 0, 0, 0, 0, 0, 0};
  __builtin_amdgcn_tensor_load_to_lds(g0, g1, z4, z4, z8, 0);
#else
  __builtin_amdgcn_tensor_load_to_lds(g0, g1, z4, z4, 0);
#endif
}

// ---------------- kernels ---------------------------------------------
__global__ void init_kernel(float* X, float* Y, float* St, float* out, int S) {
  int k = blockIdx.x * blockDim.x + threadIdx.x;
  if (k >= NSLOTS) return;
  float x = 0.f, y = 0.f, st = -1.f;
  if (k < NGRID) {
    int iy = k / NGX, ix = k - iy * NGX;
    x = (float)(MARGIN + STRIDEP * ix);
    y = (float)(MARGIN + STRIDEP * iy);
    st = 0.f;
  }
  X[k] = x; Y[k] = y; St[k] = st;
  out[(size_t)2 * k]     = x;
  out[(size_t)2 * k + 1] = y;
  out[(size_t)S * NSLOTS * 2 + k] = (st >= 0.f) ? 1.f : 0.f;
}

// prune: forward-backward check -> on mask; also zero occ for this step.
// flow_f tile staged to LDS via the Tensor Data Mover (wave 0 issues).
__global__ void prune_kernel(const float* __restrict__ ff,
                             const float* __restrict__ fb,
                             float* __restrict__ on, float* __restrict__ occ) {
  extern __shared__ float smem[];               // [2][PR_ROWS][W]
  const int y0 = blockIdx.x * PR_ROWS;
  if (threadIdx.x < 32) {                       // one wave issues both TDM loads
    tdm_load_rows((const void*)(ff + (size_t)y0 * W), 0u);
    tdm_load_rows((const void*)(ff + (size_t)HW + (size_t)y0 * W),
                  (unsigned)(PR_ROWS * W * 4));
    __builtin_amdgcn_s_wait_tensorcnt(0);
  }
  __syncthreads();
  const float* fb1 = fb + HW;
#pragma unroll
  for (int j = 0; j < (PR_ROWS * W) / 256; ++j) {
    int p = (int)threadIdx.x + 256 * j;         // 0 .. PR_ROWS*W-1
    int ly = p / W, x = p - ly * W, y = y0 + ly;
    float f0 = smem[p];
    float f1 = smem[PR_ROWS * W + p];
    Bil b = mkbil((float)x + f0, (float)y + f1);
    float b0 = b.w00 * fb[b.i00]  + b.w01 * fb[b.i01]  + b.w10 * fb[b.i10]  + b.w11 * fb[b.i11];
    float b1 = b.w00 * fb1[b.i00] + b.w01 * fb1[b.i01] + b.w10 * fb1[b.i10] + b.w11 * fb1[b.i11];
    float d0 = f0 + b0, d1 = f1 + b1;
    float diff = sqrtf(d0 * d0 + d1 * d1);
    float mag  = sqrtf(f0 * f0 + f1 * f1 + b0 * b0 + b1 * b1);
    int idx = y * W + x;
    on[idx]  = (diff > 0.01f * mag + 0.1f) ? 0.f : 1.f;
    occ[idx] = 0.f;
  }
}

// advect: move each slot along flow, validity checks, occupancy scatter.
__global__ void advect_kernel(const float* __restrict__ ff,
                              const float* __restrict__ on,
                              float* X, float* Y, float* St,
                              int* __restrict__ freeF, float* __restrict__ occ,
                              int s, int steps) {
  int k = blockIdx.x * blockDim.x + threadIdx.x;
  if (k >= NSLOTS) return;
  float Xr = X[k], Yr = Y[k], st = St[k];
  bool on_mask = (st >= 0.f);
  Bil b = mkbil(Xr, Yr);
  const float* ff1 = ff + HW;
  float u = b.w00 * ff[b.i00]  + b.w01 * ff[b.i01]  + b.w10 * ff[b.i10]  + b.w11 * ff[b.i11];
  float v = b.w00 * ff1[b.i00] + b.w01 * ff1[b.i01] + b.w10 * ff1[b.i10] + b.w11 * ff1[b.i11];
  float xt = Xr + u, yt = Yr + v;
  float fbv = b.w00 * on[b.i00] + b.w01 * on[b.i01] + b.w10 * on[b.i10] + b.w11 * on[b.i11];
  bool mchk = (xt >= (float)MARGIN) && (yt >= (float)MARGIN) &&
              (xt < (float)(W - MARGIN)) && (yt < (float)(H - MARGIN));
  bool checks = mchk && (fbv > 0.5f);
  bool is_last = (s == steps - 1);
  bool choose = on_mask && (is_last || checks);
  float Xn = choose ? xt : 0.f;
  float Yn = choose ? yt : 0.f;
  float Stn = choose ? st : -1.f;
  X[k] = Xn; Y[k] = Yn; St[k] = Stn;
  freeF[k] = (Stn < 0.f) ? 1 : 0;
  if (choose) {
    int xi = (int)fminf(fmaxf(xt, 0.f), (float)(W - 1));
    int yi = (int)fminf(fmaxf(yt, 0.f), (float)(H - 1));
    occ[yi * W + xi] = 1.f;                      // racing identical stores: fine
  }
}

// dilate+probe: added[g] = (5x5 occ window sum == 0) at integer grid pts.
// Vertical 5-sum -> 16x96 f16 A tile in LDS; horizontal 5-sum via banded
// 0/1 B matrix with three accumulated v_wmma_f32_16x16x32_f16. One wave/tile.
__global__ void dilate_kernel(const float* __restrict__ occ, int* __restrict__ added) {
  __shared__ _Float16 vs[16 * 96];
  const int tc = blockIdx.x, tr = blockIdx.y;
  const int lane = (int)threadIdx.x;             // 32 threads (1 wave)
  const int x0 = MARGIN + 64 * tc - 2;
#pragma unroll 4
  for (int j = 0; j < 48; ++j) {
    int idx = lane + 32 * j;
    int m = idx / 96, k = idx - m * 96;
    int x = x0 + k;
    float sum = 0.f;
    if (x < W) {
      int yb = MARGIN + 4 * (tr * 16 + m) - 2;
#pragma unroll
      for (int dy = 0; dy < 5; ++dy) {
        int y = yb + dy;
        if (y < H) sum += occ[y * W + x];
      }
    }
    vs[idx] = (_Float16)sum;
  }
  __syncthreads();
  v8f acc = {0.f, 0.f, 0.f, 0.f, 0.f, 0.f, 0.f, 0.f};
  const int mA = lane & 15, kA = (lane < 16) ? 0 : 8;    // A 16x32 lane map
  const int nB = lane & 15, kB = (lane < 16) ? 0 : 8;    // B 32x16 lane map
#pragma unroll
  for (int c = 0; c < 3; ++c) {
    v16h a, bm;
#pragma unroll
    for (int h = 0; h < 16; ++h) {
      int ka = ((h < 8) ? (kA + h) : (kA + 16 + h - 8)) + 32 * c;
      a[h] = vs[mA * 96 + ka];
      int kb = ((h < 8) ? (kB + h) : (kB + 16 + h - 8)) + 32 * c;
      bm[h] = (kb >= 4 * nB && kb <= 4 * nB + 4) ? (_Float16)1.0f : (_Float16)0.0f;
    }
    acc = __builtin_amdgcn_wmma_f32_16x16x32_f16(false, a, false, bm,
                                                 (short)0, acc, false, false);
  }
  const int mOff = (lane < 16) ? 0 : 8;                  // C/D 16x16 f32 map
#pragma unroll
  for (int r = 0; r < 8; ++r) {
    int gr = tr * 16 + r + mOff, gc = tc * 16 + nB;
    if (gr < NGY && gc < NGX)
      added[gr * NGX + gc] = (acc[r] == 0.f) ? 1 : 0;
  }
}

// single-block stable compaction (replaces reference argsort) + output row.
__device__ int compact_flags(const int* __restrict__ flags, int N,
                             int* __restrict__ list, int* sh) {
  int run = 0;
  for (int base = 0; base < N; base += 1024) {
    int i = base + (int)threadIdx.x;
    int f = (i < N) ? flags[i] : 0;
    sh[threadIdx.x] = f;
    __syncthreads();
    for (int off = 1; off < 1024; off <<= 1) {
      int add = ((int)threadIdx.x >= off) ? sh[threadIdx.x - off] : 0;
      __syncthreads();
      sh[threadIdx.x] += add;
      __syncthreads();
    }
    int incl = sh[threadIdx.x];
    if (f) list[run + incl - 1] = i;
    run += sh[1023];
    __syncthreads();
  }
  return run;
}

__global__ void __launch_bounds__(1024)
birth_kernel(const int* __restrict__ freeF, const int* __restrict__ added,
             float* X, float* Y, float* St,
             int* free_list, int* addg_list,
             float* __restrict__ out, int s, int S) {
  __shared__ int sh[1024];
  int nf = compact_flags(freeF, NSLOTS, free_list, sh);
  int na = compact_flags(added, NGRID, addg_list, sh);
  int m = (na < nf) ? na : nf;
  for (int r = (int)threadIdx.x; r < m; r += 1024) {
    int slot = free_list[r];
    int g = addg_list[r];
    int iy = g / NGX, ix = g - iy * NGX;
    X[slot]  = (float)(MARGIN + STRIDEP * ix);
    Y[slot]  = (float)(MARGIN + STRIDEP * iy);
    St[slot] = (float)(s + 1);
  }
  __syncthreads();
  float* outXY = out + (size_t)(s + 1) * NSLOTS * 2;
  float* outA  = out + (size_t)S * NSLOTS * 2 + (size_t)(s + 1) * NSLOTS;
  for (int k = (int)threadIdx.x; k < NSLOTS; k += 1024) {
    outXY[2 * k]     = X[k];
    outXY[2 * k + 1] = Y[k];
    outA[k] = (St[k] >= 0.f) ? 1.f : 0.f;
  }
}

// ---------------- host side -------------------------------------------
extern "C" void kernel_launch(void* const* d_in, const int* in_sizes, int n_in,
                              void* d_out, int out_size, void* d_ws, size_t ws_size,
                              hipStream_t stream) {
  (void)n_in; (void)out_size; (void)ws_size;
  const float* flows_f = (const float*)d_in[1];  // [S-1,2,H,W]
  const float* flows_b = (const float*)d_in[2];  // [S-1,2,H,W]
  float* out = (float*)d_out;                    // [S,NSLOTS,2] ++ [S,NSLOTS]
  const int steps = in_sizes[1] / (2 * HW);      // S-1
  const int S = steps + 1;

  // workspace layout
  float* X     = (float*)d_ws;
  float* Y     = X + NSLOTS;
  float* St    = Y + NSLOTS;
  int*   freeF = (int*)(St + NSLOTS);
  float* onM   = (float*)(freeF + NSLOTS);
  float* occ   = onM + HW;
  int*   added = (int*)(occ + HW);
  int*   free_list = added + NGRID;
  int*   addg_list = free_list + NSLOTS;

  init_kernel<<<(NSLOTS + 255) / 256, 256, 0, stream>>>(X, Y, St, out, S);
  for (int s = 0; s < steps; ++s) {
    const float* ff = flows_f + (size_t)s * 2 * HW;
    const float* fb = flows_b + (size_t)s * 2 * HW;
    prune_kernel<<<H / PR_ROWS, 256, 2 * PR_ROWS * W * sizeof(float), stream>>>(
        ff, fb, onM, occ);
    advect_kernel<<<(NSLOTS + 255) / 256, 256, 0, stream>>>(
        ff, onM, X, Y, St, freeF, occ, s, steps);
    dilate_kernel<<<dim3((NGX + 15) / 16, (NGY + 15) / 16), 32, 0, stream>>>(occ, added);
    birth_kernel<<<1, 1024, 0, stream>>>(freeF, added, X, Y, St,
                                         free_list, addg_list, out, s, S);
  }
}
